// Multi_head_self_attention_44143673868503
// MI455X (gfx1250) — matrix operand order repaired and checked by reading, not compiled, not run on hardware
//
#include <hip/hip_runtime.h>

#define HEADS 16
#define DHEAD 64
#define SDIM  1024
#define LDIM  2048
#define BATCH 4

#define USE_ASYNC_LDS 1   // global_load_async_to_lds_b128 staging (fallback: plain load+ds_store)

typedef __attribute__((ext_vector_type(16))) __bf16 v16bf;
typedef __attribute__((ext_vector_type(8)))  float  v8f;
typedef unsigned int uint32x4 __attribute__((ext_vector_type(4)));
typedef float        f32x4    __attribute__((ext_vector_type(4)));

union FragB16 {
    v16bf    v;
    uint32x4 q[2];
};

// float -> bf16 bits, round-to-nearest-even
__device__ inline unsigned short f2bf(float f) {
    unsigned int u = __float_as_uint(f);
    u += 0x7FFFu + ((u >> 16) & 1u);
    return (unsigned short)(u >> 16);
}

// ---- WMMA fragment loaders (bf16, 16x16x32) -------------------------------
// A 16x32 (MxK) row-major, base at (row0,k0):
//   lanes 0-15: row=M, halves 0-7 = K0..7, 8-15 = K16..23; lanes 16-31: K+8.
__device__ inline v16bf load_frag_a(const unsigned short* base, int stride, int lane) {
    const unsigned short* p = base + (size_t)(lane & 15) * stride + ((lane & 16) ? 8 : 0);
    FragB16 f;
    f.q[0] = *(const uint32x4*)(p);
    f.q[1] = *(const uint32x4*)(p + 16);
    return f.v;
}

// B 32x16 (KxN) stored N-major: base at (col0,k0); lanes 0-15: K0..7, K16..23; 16-31: K+8.
__device__ inline v16bf load_frag_b(const unsigned short* base, int stride, int lane) {
    const unsigned short* p = base + (size_t)(lane & 15) * stride + ((lane & 16) ? 8 : 0);
    FragB16 f;
    f.q[0] = *(const uint32x4*)(p);
    f.q[1] = *(const uint32x4*)(p + 16);
    return f.v;
}

__device__ inline v8f wmma_bf16(v16bf a, v16bf b, v8f c) {
    return __builtin_amdgcn_wmma_f32_16x16x32_bf16(false, a, false, b, (short)0, c, false, false);
}

// ---- 16-byte global->LDS staging ------------------------------------------
__device__ inline void stage16(unsigned short* lds_dst, const unsigned short* g) {
#if USE_ASYNC_LDS
    unsigned lds_off = (unsigned)(uintptr_t)lds_dst;            // addr[31:0] = LDS offset
    unsigned long long ga = (unsigned long long)(uintptr_t)g;
    asm volatile("global_load_async_to_lds_b128 %0, %1, off"
                 :: "v"(lds_off), "v"(ga) : "memory");
#else
    *(uint32x4*)lds_dst = *(const uint32x4*)g;
#endif
}

__device__ inline void stage_wait() {
#if USE_ASYNC_LDS
    asm volatile("s_wait_asynccnt 0" ::: "memory");
#endif
}

// ---- Kernel: fp32 -> bf16 bits (flat) -------------------------------------
__global__ __launch_bounds__(256) void cvt_bf16_kernel(const float* __restrict__ x,
                                                       unsigned short* __restrict__ xb, int n) {
    int i = (blockIdx.x * 256 + threadIdx.x) * 4;
    if (i + 3 < n) {
        f32x4 f = *(const f32x4*)(x + i);
        xb[i + 0] = f2bf(f.x);
        xb[i + 1] = f2bf(f.y);
        xb[i + 2] = f2bf(f.z);
        xb[i + 3] = f2bf(f.w);
    }
}

// ---- Kernel: x (b,c,l) fp32 -> xT (b,l,c) bf16 via LDS tile ---------------
__global__ __launch_bounds__(256) void transpose_bf16_kernel(const float* __restrict__ x,
                                                             unsigned short* __restrict__ xT) {
    __shared__ float tile[32][33];
    const int b = blockIdx.z;
    const int l0 = blockIdx.x * 32, c0 = blockIdx.y * 32;
    const int tx = threadIdx.x & 31, ty = threadIdx.x >> 5;   // 32 x 8
    const float* xb = x + (size_t)b * SDIM * LDIM;
    #pragma unroll
    for (int i = 0; i < 32; i += 8)
        tile[ty + i][tx] = xb[(size_t)(c0 + ty + i) * LDIM + l0 + tx];
    __syncthreads();
    unsigned short* xTb = xT + (size_t)b * LDIM * SDIM;
    #pragma unroll
    for (int i = 0; i < 32; i += 8)
        xTb[(size_t)(l0 + ty + i) * SDIM + c0 + tx] = f2bf(tile[tx][ty + i]);
}

// ---- Kernel: conv1x1 GEMM, all-bf16 inputs, double-buffered async pipeline -
// out[b] = W (MxK bf16 row-major) @ X[b] (XT: N-major NxK bf16) + bias
// MODE 0: store bf16 per head -> out[(b,h,l,d)]   (K/Q)
// MODE 1: store fp32 natural  -> out[(b,o,l)]     (final projection)
// Block: 256 thr = 8 waves (2x4), tile 64(M) x 128(N), K step 32.
template <int MODE>
__global__ __launch_bounds__(256) void gemm_bf16_kernel(const unsigned short* __restrict__ Wb,
                                                        const float* __restrict__ bias,
                                                        const unsigned short* __restrict__ XT,
                                                        void* __restrict__ out) {
    __shared__ __attribute__((aligned(16))) unsigned short Alds[2][64 * 40];
    __shared__ __attribute__((aligned(16))) unsigned short Blds[2][128 * 40];

    const int tid = threadIdx.x;
    const int b  = blockIdx.z;
    const int m0 = blockIdx.y * 64;
    const int n0 = blockIdx.x * 128;
    const unsigned short* XTb = XT + (size_t)b * LDIM * SDIM;

    const int wave = tid >> 5, lane = tid & 31;
    const int wm = (wave >> 2) * 32;  // 0 / 32
    const int wn = (wave & 3)  * 32;  // 0..96

    // per-thread staging slots (contiguous 16B pieces)
    const int arow = tid >> 2, akk = (tid & 3) * 8;              // A: 64x32, 1 op
    const int brow0 = tid >> 1, bkk0 = (tid & 1) * 16;           // B: 128x32, 2 ops (8+8)

    const unsigned short* gA  = Wb  + (size_t)(m0 + arow) * SDIM + akk;
    const unsigned short* gB  = XTb + (size_t)(n0 + brow0) * SDIM + bkk0;

    v8f acc[2][2] = {};

    // prologue: issue tile 0 into buffer 0
    stage16(&Alds[0][arow * 40 + akk], gA);
    stage16(&Blds[0][brow0 * 40 + bkk0],     gB);
    stage16(&Blds[0][brow0 * 40 + bkk0 + 8], gB + 8);

    int p = 0;
    for (int k0 = 0; k0 < SDIM; k0 += 32, p ^= 1) {
        stage_wait();        // this wave's async writes (tile k0) have landed
        __syncthreads();     // all waves' writes visible; buffer p^1 free to refill

        if (k0 + 32 < SDIM) {
            stage16(&Alds[p ^ 1][arow * 40 + akk], gA + k0 + 32);
            stage16(&Blds[p ^ 1][brow0 * 40 + bkk0],     gB + k0 + 32);
            stage16(&Blds[p ^ 1][brow0 * 40 + bkk0 + 8], gB + k0 + 40);
        }

        v16bf a0  = load_frag_a(&Alds[p][(wm +  0) * 40], 40, lane);
        v16bf a1  = load_frag_a(&Alds[p][(wm + 16) * 40], 40, lane);
        v16bf b0v = load_frag_b(&Blds[p][(wn +  0) * 40], 40, lane);
        v16bf b1v = load_frag_b(&Blds[p][(wn + 16) * 40], 40, lane);
        acc[0][0] = wmma_bf16(a0, b0v, acc[0][0]);
        acc[0][1] = wmma_bf16(a0, b1v, acc[0][1]);
        acc[1][0] = wmma_bf16(a1, b0v, acc[1][0]);
        acc[1][1] = wmma_bf16(a1, b1v, acc[1][1]);
    }

    const int cn  = lane & 15;
    const int lhi = (lane & 16) ? 8 : 0;
    #pragma unroll
    for (int sm = 0; sm < 2; ++sm)
        #pragma unroll
        for (int sn = 0; sn < 2; ++sn)
            #pragma unroll
            for (int r = 0; r < 8; ++r) {
                int grow = m0 + wm + sm * 16 + r + lhi;
                int gcol = n0 + wn + sn * 16 + cn;
                float v = acc[sm][sn][r] + bias[grow];
                if (MODE == 0) {
                    unsigned short* o = (unsigned short*)out;
                    int h = grow >> 6, d = grow & 63;
                    o[(((size_t)b * HEADS + h) * LDIM + gcol) * DHEAD + d] = f2bf(v);
                } else {
                    float* o = (float*)out;
                    o[((size_t)b * SDIM + grow) * LDIM + gcol] = v;
                }
            }
}

// ---- Kernel: softmax stats (per-row running max / sum-exp) ----------------
__global__ __launch_bounds__(256) void attn_stats_kernel(const unsigned short* __restrict__ Kt,
                                                         const unsigned short* __restrict__ Qt,
                                                         float* __restrict__ mbuf,
                                                         float* __restrict__ zbuf) {
    const int wave = threadIdx.x >> 5, lane = threadIdx.x & 31;
    const int b = blockIdx.z, h = blockIdx.y;
    const int i0 = blockIdx.x * 128 + wave * 16;
    const size_t base = ((size_t)b * HEADS + h) * (size_t)LDIM * DHEAD;
    const unsigned short* Kbh = Kt + base;
    const unsigned short* Qbh = Qt + base;
    const float scale = 2.0f / (float)LDIM;

    v16bf a0 = load_frag_a(Kbh + (size_t)i0 * DHEAD +  0, DHEAD, lane);
    v16bf a1 = load_frag_a(Kbh + (size_t)i0 * DHEAD + 32, DHEAD, lane);

    float M[8], Ssum[8];
    #pragma unroll
    for (int r = 0; r < 8; ++r) { M[r] = -3.0e38f; Ssum[r] = 0.0f; }

    for (int j0 = 0; j0 < LDIM; j0 += 16) {
        v16bf q0 = load_frag_b(Qbh + (size_t)j0 * DHEAD +  0, DHEAD, lane);
        v16bf q1 = load_frag_b(Qbh + (size_t)j0 * DHEAD + 32, DHEAD, lane);
        v8f c = {};
        c = wmma_bf16(a0, q0, c);
        c = wmma_bf16(a1, q1, c);
        #pragma unroll
        for (int r = 0; r < 8; ++r) {
            float s  = c[r] * scale;
            float nm = fmaxf(M[r], s);
            Ssum[r]  = Ssum[r] * __expf(M[r] - nm) + __expf(s - nm);
            M[r]     = nm;
        }
    }
    #pragma unroll
    for (int mask = 1; mask < 16; mask <<= 1) {
        #pragma unroll
        for (int r = 0; r < 8; ++r) {
            float om = __shfl_xor(M[r], mask, 16);
            float os = __shfl_xor(Ssum[r], mask, 16);
            float nm = fmaxf(M[r], om);
            Ssum[r]  = Ssum[r] * __expf(M[r] - nm) + os * __expf(om - nm);
            M[r]     = nm;
        }
    }
    if ((lane & 15) == 0) {
        int roff = (lane & 16) ? 8 : 0;
        size_t sb = ((size_t)b * HEADS + h) * LDIM + i0 + roff;
        #pragma unroll
        for (int r = 0; r < 8; ++r) {
            mbuf[sb + r] = M[r];
            zbuf[sb + r] = 1.0f / Ssum[r];
        }
    }
}

// ---- Kernel: out(d,j) = sum_i x(d,i)*exp(s(i,j)-m_i)/Z_i ------------------
// Writes bf16 output transposed -> obufT[(b,l,c)], directly consumable as the
// N-major B operand of the final projection GEMM.
__global__ __launch_bounds__(256) void attn_apply_kernel(const unsigned short* __restrict__ Kt,
                                                         const unsigned short* __restrict__ Qt,
                                                         const unsigned short* __restrict__ Xv,
                                                         const float* __restrict__ mbuf,
                                                         const float* __restrict__ zbuf,
                                                         unsigned short* __restrict__ obufT) {
    __shared__ __attribute__((aligned(16))) float ldsM[32];
    __shared__ __attribute__((aligned(16))) float ldsZ[32];
    __shared__ __attribute__((aligned(16))) unsigned short Pl[8 * 16 * 40];

    const int tid = threadIdx.x, wave = tid >> 5, lane = tid & 31;
    const int b = blockIdx.z, h = blockIdx.y;
    const int j0 = blockIdx.x * 128 + wave * 16;
    const size_t base = ((size_t)b * HEADS + h) * (size_t)LDIM * DHEAD;
    const unsigned short* Kbh = Kt + base;
    const unsigned short* Qbh = Qt + base;
    const unsigned short* Xbh = Xv + ((size_t)b * SDIM + h * DHEAD) * LDIM;
    const size_t statbase = ((size_t)b * HEADS + h) * LDIM;
    const float scale = 2.0f / (float)LDIM;

    v16bf q0 = load_frag_b(Qbh + (size_t)j0 * DHEAD +  0, DHEAD, lane);
    v16bf q1 = load_frag_b(Qbh + (size_t)j0 * DHEAD + 32, DHEAD, lane);

    v8f acc[4] = {};
    unsigned short* Pw = &Pl[wave * 16 * 40];
    const int cn  = lane & 15;
    const int lhi = (lane & 16) ? 8 : 0;

    for (int i0 = 0; i0 < LDIM; i0 += 32) {
        if (tid < 32) {
            ldsM[tid] = mbuf[statbase + i0 + tid];
            ldsZ[tid] = zbuf[statbase + i0 + tid];
        }
        __syncthreads();

        #pragma unroll
        for (int isub = 0; isub < 32; isub += 16) {
            v16bf a0 = load_frag_a(Kbh + (size_t)(i0 + isub) * DHEAD +  0, DHEAD, lane);
            v16bf a1 = load_frag_a(Kbh + (size_t)(i0 + isub) * DHEAD + 32, DHEAD, lane);
            v8f c = {};
            c = wmma_bf16(a0, q0, c);
            c = wmma_bf16(a1, q1, c);
            #pragma unroll
            for (int r = 0; r < 8; ++r) {
                int il = isub + r + lhi;
                float p = __expf(c[r] * scale - ldsM[il]) * ldsZ[il];
                Pw[cn * 40 + il] = f2bf(p);
            }
        }
        asm volatile("s_wait_dscnt 0" ::: "memory");  // wave-local LDS RAW (cross-lane)

        v16bf pb = load_frag_b(Pw, 40, lane);
        #pragma unroll
        for (int t = 0; t < 4; ++t) {
            v16bf ax = load_frag_a(Xbh + (size_t)(t * 16) * LDIM + i0, LDIM, lane);
            acc[t] = wmma_bf16(ax, pb, acc[t]);
        }
        __syncthreads();
    }

    // packed bf16 stores, transposed layout (b, l, c): 16B per tile per lane
    #pragma unroll
    for (int t = 0; t < 4; ++t) {
        union { unsigned short u[8]; uint32x4 q; } pk;
        #pragma unroll
        for (int r = 0; r < 8; ++r) pk.u[r] = f2bf(acc[t][r]);
        size_t off = ((size_t)b * LDIM + j0 + cn) * SDIM + h * DHEAD + t * 16 + lhi;
        *(uint32x4*)&obufT[off] = pk.q;
    }
}

// ---------------------------------------------------------------------------
extern "C" void kernel_launch(void* const* d_in, const int* in_sizes, int n_in,
                              void* d_out, int out_size, void* d_ws, size_t ws_size,
                              hipStream_t stream) {
    const float* x  = (const float*)d_in[0];
    const float* Wk = (const float*)d_in[1];
    const float* bk = (const float*)d_in[2];
    const float* Wq = (const float*)d_in[3];
    const float* bq = (const float*)d_in[4];
    const float* Wp = (const float*)d_in[5];
    const float* bp = (const float*)d_in[6];

    // workspace layout (~87 MB)
    char* ws = (char*)d_ws;
    unsigned short* Kt    = (unsigned short*)(ws);                       // 16 MB (b,h,l,d)
    unsigned short* Qt    = (unsigned short*)(ws + ((size_t)16 << 20));  // 16 MB
    unsigned short* xbf   = (unsigned short*)(ws + ((size_t)32 << 20));  // 16 MB (b,c,l)
    unsigned short* xbfT  = (unsigned short*)(ws + ((size_t)48 << 20));  // 16 MB (b,l,c)
    unsigned short* obufT = (unsigned short*)(ws + ((size_t)64 << 20));  // 16 MB (b,l,c)
    float*          mbuf  = (float*)(ws + ((size_t)80 << 20));           // 512 KB
    float*          zbuf  = (float*)(ws + ((size_t)80 << 20) + ((size_t)512 << 10));
    unsigned short* Wkb   = (unsigned short*)(ws + ((size_t)81 << 20));  // 2 MB each
    unsigned short* Wqb   = (unsigned short*)(ws + ((size_t)83 << 20));
    unsigned short* Wpb   = (unsigned short*)(ws + ((size_t)85 << 20));

    const int n  = BATCH * SDIM * LDIM;
    const int nw = SDIM * SDIM;
    cvt_bf16_kernel<<<dim3(n / 1024),  dim3(256), 0, stream>>>(x,  xbf, n);
    cvt_bf16_kernel<<<dim3(nw / 1024), dim3(256), 0, stream>>>(Wk, Wkb, nw);
    cvt_bf16_kernel<<<dim3(nw / 1024), dim3(256), 0, stream>>>(Wq, Wqb, nw);
    cvt_bf16_kernel<<<dim3(nw / 1024), dim3(256), 0, stream>>>(Wp, Wpb, nw);
    transpose_bf16_kernel<<<dim3(LDIM / 32, SDIM / 32, BATCH), dim3(256), 0, stream>>>(x, xbfT);

    dim3 ggrid(LDIM / 128, SDIM / 64, BATCH);
    gemm_bf16_kernel<0><<<ggrid, dim3(256), 0, stream>>>(Wkb, bk, xbfT, (void*)Kt);
    gemm_bf16_kernel<0><<<ggrid, dim3(256), 0, stream>>>(Wqb, bq, xbfT, (void*)Qt);

    dim3 agrid(LDIM / 128, HEADS, BATCH);
    attn_stats_kernel<<<agrid, dim3(256), 0, stream>>>(Kt, Qt, mbuf, zbuf);
    attn_apply_kernel<<<agrid, dim3(256), 0, stream>>>(Kt, Qt, xbf, mbuf, zbuf, obufT);

    gemm_bf16_kernel<1><<<ggrid, dim3(256), 0, stream>>>(Wpb, bp, obufT, d_out);
}
